// HypergraphViewGenerator_13778255086165
// MI455X (gfx1250) — matrix operand order repaired and checked by reading, not compiled, not run on hardware
//
#include <hip/hip_runtime.h>
#include <hip/hip_bf16.h>
#include <math.h>

// ---------------------------------------------------------------------------
// HypergraphViewGenerator for MI455X (gfx1250, wave32).
//  - X@W1 (100000x128x64): f16 WMMA. X pre-converted to f16 so A fragments are
//    2x global_load_b128; W1 transposed->f16 in LDS so B fragments are single
//    32B LDS reads, hoisted to registers and reused across 4 M-tiles/wave.
//  - segment sums via f32 global atomics; all accumulators are L2-resident
//    (192 MB L2 >> 56 MB accumulator state). Binv row-scaling fused into the
//    gather side of the following scatter (bit-identical per element).
// ---------------------------------------------------------------------------

typedef __attribute__((ext_vector_type(16))) _Float16 v16h;
typedef __attribute__((ext_vector_type(8)))  _Float16 v8h;
typedef __attribute__((ext_vector_type(4)))  _Float16 v4h;
typedef __attribute__((ext_vector_type(8)))  float    v8f;

#define N_NODES_  100000
#define N_HEDGES_ 100000
#define N_INC_    1000000
#define IN_DIM_   128
#define HID_      64
#define EPS_      1e-10f

// ---------------- zero fill ----------------
__global__ __launch_bounds__(256) void zero_f32_k(float* __restrict__ p, size_t n) {
  size_t t = (size_t)blockIdx.x * blockDim.x + threadIdx.x;
  if (t < n) p[t] = 0.0f;
}

// ---------------- degree counts (+ index-stream prefetch) ----------------
__global__ __launch_bounds__(256) void degrees_k(const int* __restrict__ ei0,
                                                 const int* __restrict__ ei1,
                                                 float* __restrict__ Ddeg,
                                                 float* __restrict__ Bdeg) {
  size_t i = (size_t)blockIdx.x * blockDim.x + threadIdx.x;
  if (i >= (size_t)N_INC_) return;
  if (i + 16384 < (size_t)N_INC_) {        // stream-ahead hint -> global_prefetch_b8
    __builtin_prefetch(ei0 + i + 16384, 0, 0);
    __builtin_prefetch(ei1 + i + 16384, 0, 0);
  }
  atomicAdd(Ddeg + ei0[i], 1.0f);
  atomicAdd(Bdeg + ei1[i], 1.0f);
}

// ---------------- f32 -> f16 conversion (float4 -> 4xf16) ----------------
__global__ __launch_bounds__(256) void cvt_f16_k(const float* __restrict__ src,
                                                 _Float16* __restrict__ dst,
                                                 size_t n4) {
  size_t t = (size_t)blockIdx.x * blockDim.x + threadIdx.x;
  if (t >= n4) return;
  float4 v = ((const float4*)src)[t];
  v4h o = { (_Float16)v.x, (_Float16)v.y, (_Float16)v.z, (_Float16)v.w };
  *(v4h*)(dst + t * 4) = o;
}

// ---------------- Xt = X @ W1 via WMMA ----------------
// 8 waves/block; W1^T (f16, 64x128) staged in LDS once per block.
// Each wave: loads 16 B fragments (one 32B LDS read each) into registers,
// then computes 4 M-tiles of 16x64, A fragments as 2x global_load_b128.
__global__ __launch_bounds__(256) void gemm_xw1_k(const _Float16* __restrict__ Xh,
                                                  const float* __restrict__ W1,
                                                  float* __restrict__ Xt) {
  __shared__ __align__(32) _Float16 Wt[HID_ * IN_DIM_];   // [n][k], 16 KB

  {
    int t = threadIdx.x;
#pragma unroll
    for (int j = 0; j < 32; ++j) {
      int e = t * 32 + j;
      int n = e >> 7, k = e & (IN_DIM_ - 1);
      Wt[e] = (_Float16)W1[(size_t)k * HID_ + n];
    }
  }
  __syncthreads();

  const int  lane = threadIdx.x & 31;
  const int  wid  = blockIdx.x * 8 + (threadIdx.x >> 5);
  const bool hi   = lane >= 16;
  const int  l    = lane & 15;

  // B fragments: lane = column N (l); lanes 0-15 hold K=ks..ks+15,
  // lanes 16-31 hold K=ks+16..ks+31 -> contiguous 16 halfs in Wt.
  v16h bf[16];
#pragma unroll
  for (int ks = 0; ks < 4; ++ks)
#pragma unroll
    for (int nt = 0; nt < 4; ++nt) {
      const _Float16* p = Wt + (size_t)(nt * 16 + l) * IN_DIM_ + ks * 32 + (hi ? 16 : 0);
      bf[ks * 4 + nt] = *(const v16h*)p;
    }

  for (int mt = 0; mt < 4; ++mt) {
    int tile = wid * 4 + mt;
    if (tile >= N_NODES_ / 16) break;            // wave-uniform: EXEC stays all-1s
    const _Float16* __restrict__ Arow = Xh + (size_t)(tile * 16 + l) * IN_DIM_;
    v8f c[4] = {};
#pragma unroll
    for (int ks = 0; ks < 4; ++ks) {
      // A fragment: lane holds K=ks+{0..15}(+16 if hi), contiguous like B
      v8h alo = *(const v8h*)(Arow + ks * 32 + (hi ? 16 : 0));
      v8h ahi = *(const v8h*)(Arow + ks * 32 + 8 + (hi ? 16 : 0));
      v16h a = __builtin_shufflevector(alo, ahi,
                                       0, 1, 2, 3, 4, 5, 6, 7,
                                       8, 9, 10, 11, 12, 13, 14, 15);
#pragma unroll
      for (int nt = 0; nt < 4; ++nt)
        c[nt] = __builtin_amdgcn_wmma_f32_16x16x32_f16(
            /*neg_a=*/false, a, /*neg_b=*/false, bf[ks * 4 + nt],
            /*c_mod=*/(short)0, c[nt], /*reuse_a=*/false, /*reuse_b=*/false);
    }
    // C layout: VGPR r -> row tile*16 + r + (hi?8:0), col = nt*16 + l
    const int srow = tile * 16 + (hi ? 8 : 0);
#pragma unroll
    for (int r = 0; r < 8; ++r) {
      float* __restrict__ o = Xt + (size_t)(srow + r) * HID_ + l;
      o[0]  = c[0][r];
      o[16] = c[1][r];
      o[32] = c[2][r];
      o[48] = c[3][r];
    }
  }
}

// ---- 64-wide segment-sum: dst[sidx[i]] += src[gidx[i]] * inv(gdeg[gidx[i]]) ----
__global__ __launch_bounds__(256) void scatter64_k(const int* __restrict__ gidx,
                                                   const int* __restrict__ sidx,
                                                   const float* __restrict__ src,
                                                   const float* __restrict__ gdeg,
                                                   float* __restrict__ dst) {
  size_t t = (size_t)blockIdx.x * blockDim.x + threadIdx.x;
  size_t i = t >> 4;
  if (i >= (size_t)N_INC_) return;
  int f = (int)(t & 15) * 4;
  int g = gidx[i], s = sidx[i];
  float sc = 1.0f;
  if (gdeg) {
    float d = gdeg[g];
    sc = (d > 0.0f) ? (1.0f / d) : 0.0f;
  }
  const float4 v = *(const float4*)(src + (size_t)g * HID_ + f);
  float* d = dst + (size_t)s * HID_ + f;
  atomicAdd(d + 0, v.x * sc);
  atomicAdd(d + 1, v.y * sc);
  atomicAdd(d + 2, v.z * sc);
  atomicAdd(d + 3, v.w * sc);
}

// ---------------- x[r,:] = x[r,:]*inv(deg[r]) + bias[f] ----------------
__global__ __launch_bounds__(256) void finish64_k(float* __restrict__ x,
                                                  const float* __restrict__ deg,
                                                  const float* __restrict__ bias,
                                                  size_t nrows) {
  size_t t = (size_t)blockIdx.x * blockDim.x + threadIdx.x;
  if (t >= nrows * HID_) return;
  float d = deg[t >> 6];
  float inv = (d > 0.0f) ? (1.0f / d) : 0.0f;
  x[t] = x[t] * inv + bias[t & (HID_ - 1)];
}

// ---------------- Xt2 = H1 @ W2  (N=3 GEMV per node) ----------------
__global__ __launch_bounds__(256) void gemv3_k(const float* __restrict__ H1,
                                               const float* __restrict__ W2,
                                               float* __restrict__ out) {
  int nid = blockIdx.x * blockDim.x + threadIdx.x;
  if (nid >= N_NODES_) return;
  const float* __restrict__ h = H1 + (size_t)nid * HID_;
  float a0 = 0.f, a1 = 0.f, a2 = 0.f;
#pragma unroll 8
  for (int f = 0; f < HID_; ++f) {
    float v = h[f];
    a0 += v * W2[f * 3 + 0];
    a1 += v * W2[f * 3 + 1];
    a2 += v * W2[f * 3 + 2];
  }
  out[(size_t)nid * 3 + 0] = a0;
  out[(size_t)nid * 3 + 1] = a1;
  out[(size_t)nid * 3 + 2] = a2;
}

// ---------------- 3-wide segment-sum with optional fused gather-side scale ----
__global__ __launch_bounds__(256) void scatter3_k(const int* __restrict__ gidx,
                                                  const int* __restrict__ sidx,
                                                  const float* __restrict__ src,
                                                  const float* __restrict__ gdeg,
                                                  float* __restrict__ dst) {
  size_t i = (size_t)blockIdx.x * blockDim.x + threadIdx.x;
  if (i >= (size_t)N_INC_) return;
  int g = gidx[i], s = sidx[i];
  float sc = 1.0f;
  if (gdeg) {
    float d = gdeg[g];
    sc = (d > 0.0f) ? (1.0f / d) : 0.0f;
  }
  const float* v = src + (size_t)g * 3;
  float* d = dst + (size_t)s * 3;
  atomicAdd(d + 0, v[0] * sc);
  atomicAdd(d + 1, v[1] * sc);
  atomicAdd(d + 2, v[2] * sc);
}

// ---------------- x[r,:3] = x[r,:3]*inv(deg[r]) + b[:3] ----------------
__global__ __launch_bounds__(256) void finish3_k(float* __restrict__ x,
                                                 const float* __restrict__ deg,
                                                 const float* __restrict__ bias,
                                                 int nrows) {
  int r = blockIdx.x * blockDim.x + threadIdx.x;
  if (r >= nrows) return;
  float d = deg[r];
  float inv = (d > 0.0f) ? (1.0f / d) : 0.0f;
  x[(size_t)r * 3 + 0] = x[(size_t)r * 3 + 0] * inv + bias[0];
  x[(size_t)r * 3 + 1] = x[(size_t)r * 3 + 1] * inv + bias[1];
  x[(size_t)r * 3 + 2] = x[(size_t)r * 3 + 2] * inv + bias[2];
}

// ---------------- per-hyperedge: mean -> sigmoid -> rownorm -> gumbel argmax --
__global__ __launch_bounds__(256) void hedge_final_k(const float* __restrict__ XeSum,
                                                     const float* __restrict__ Bdeg,
                                                     const float* __restrict__ gu,
                                                     int* __restrict__ hard_idx) {
  int h = blockIdx.x * blockDim.x + threadIdx.x;
  if (h >= N_HEDGES_) return;
  float cnt = Bdeg[h];
  float inv = (cnt > 0.0f) ? (1.0f / cnt) : 0.0f;
  float x0 = 1.0f / (1.0f + __expf(-XeSum[(size_t)h * 3 + 0] * inv));
  float x1 = 1.0f / (1.0f + __expf(-XeSum[(size_t)h * 3 + 1] * inv));
  float x2 = 1.0f / (1.0f + __expf(-XeSum[(size_t)h * 3 + 2] * inv));
  float rs = x0 + x1 + x2;
  float rinv = (rs != 0.0f) ? (1.0f / rs) : 0.0f;
  x0 *= rinv; x1 *= rinv; x2 *= rinv;
  // gumbel noise; argmax(softmax(x+g)) == argmax(x+g)
  float g0 = -__logf(-__logf(gu[(size_t)h * 3 + 0] + EPS_) + EPS_);
  float g1 = -__logf(-__logf(gu[(size_t)h * 3 + 1] + EPS_) + EPS_);
  float g2 = -__logf(-__logf(gu[(size_t)h * 3 + 2] + EPS_) + EPS_);
  float l0 = x0 + g0, l1 = x1 + g1, l2 = x2 + g2;
  int idx = 0;
  float best = l0;
  if (l1 > best) { best = l1; idx = 1; }
  if (l2 > best) { idx = 2; }
  hard_idx[h] = idx;
}

// ---------------- final per-incidence sample + masked edge_index ----------------
__global__ __launch_bounds__(256) void final_out_k(const int* __restrict__ ei0,
                                                   const int* __restrict__ ei1,
                                                   const float* __restrict__ ov,
                                                   const int* __restrict__ hard_idx,
                                                   float* __restrict__ out_sample,
                                                   float* __restrict__ out_ei) {
  size_t i = (size_t)blockIdx.x * blockDim.x + threadIdx.x;
  if (i >= (size_t)N_INC_) return;
  int n = ei0[i], h = ei1[i];
  int idx = hard_idx[h];
  float reserve = (idx == 0) ? 1.0f : 0.0f;
  float maskf   = (idx == 2 && ov[n] < 0.5f) ? 1.0f : 0.0f;
  float sample  = fmaxf(reserve, maskf);
  out_sample[i] = sample;
  bool keep = sample > 0.0f;
  out_ei[i]                  = keep ? (float)n : -1.0f;
  out_ei[(size_t)N_INC_ + i] = keep ? (float)h : -1.0f;
}

// ---------------------------------------------------------------------------
extern "C" void kernel_launch(void* const* d_in, const int* in_sizes, int n_in,
                              void* d_out, int out_size, void* d_ws, size_t ws_size,
                              hipStream_t stream) {
  (void)in_sizes; (void)n_in; (void)out_size; (void)ws_size;

  const float* X  = (const float*)d_in[0];   // [100000,128]
  const float* W1 = (const float*)d_in[1];   // [128,64]
  const float* b1 = (const float*)d_in[2];   // [64]
  const float* W2 = (const float*)d_in[3];   // [64,3]
  const float* b2 = (const float*)d_in[4];   // [3]
  const float* ov = (const float*)d_in[5];   // [100000]
  const float* gu = (const float*)d_in[6];   // [100000,3]
  const int*   ei = (const int*)d_in[7];     // [2,1000000]
  const int* ei0 = ei;
  const int* ei1 = ei + N_INC_;

  // workspace layout (float units)
  float* ws = (float*)d_ws;
  size_t off = 0;
  float* Bdeg = ws + off; off += N_HEDGES_;
  float* Ddeg = ws + off; off += N_NODES_;
  float* Xe1  = ws + off; off += (size_t)N_HEDGES_ * HID_;
  float* H1   = ws + off; off += (size_t)N_NODES_ * HID_;
  float* Xe2  = ws + off; off += (size_t)N_HEDGES_ * 3;
  float* H2   = ws + off; off += (size_t)N_NODES_ * 3;
  float* XeM  = ws + off; off += (size_t)N_HEDGES_ * 3;
  const size_t zeroN = off;                                   // accumulators
  _Float16* Xh = (_Float16*)(ws + off); off += ((size_t)N_NODES_ * IN_DIM_) / 2;
  float* Xt   = ws + off; off += (size_t)N_NODES_ * HID_;     // direct-write
  float* Xt2  = ws + off; off += (size_t)N_NODES_ * 3;        // direct-write
  int*   hard = (int*)(ws + off);

  float* out_sample = (float*)d_out;                 // [1000000]
  float* out_ei     = (float*)d_out + N_INC_;        // [2,1000000]

  auto cdiv = [](size_t a, size_t b) { return (unsigned)((a + b - 1) / b); };

  // 1) clear accumulators
  zero_f32_k<<<cdiv(zeroN, 256), 256, 0, stream>>>(ws, zeroN);
  // 2) node / hyperedge degrees
  degrees_k<<<cdiv(N_INC_, 256), 256, 0, stream>>>(ei0, ei1, Ddeg, Bdeg);
  // 3) X -> f16
  cvt_f16_k<<<cdiv(((size_t)N_NODES_ * IN_DIM_) / 4, 256), 256, 0, stream>>>(
      X, Xh, ((size_t)N_NODES_ * IN_DIM_) / 4);
  // 4) Xt = X @ W1 (WMMA); 4 M-tiles per wave, 8 waves per block
  {
    unsigned waves  = cdiv(N_NODES_ / 16, 4);
    unsigned blocks = cdiv(waves, 8);
    gemm_xw1_k<<<blocks, 256, 0, stream>>>(Xh, W1, Xt);
  }
  // 5) Xe1 = segsum_h(Xt[n])
  scatter64_k<<<cdiv((size_t)N_INC_ * 16, 256), 256, 0, stream>>>(ei0, ei1, Xt, nullptr, Xe1);
  // 6) H1 = segsum_n(Xe1[h] * Binv[h]); then * 1/D + b1
  scatter64_k<<<cdiv((size_t)N_INC_ * 16, 256), 256, 0, stream>>>(ei1, ei0, Xe1, Bdeg, H1);
  finish64_k<<<cdiv((size_t)N_NODES_ * HID_, 256), 256, 0, stream>>>(H1, Ddeg, b1, N_NODES_);
  // 7) Xt2 = H1 @ W2
  gemv3_k<<<cdiv(N_NODES_, 256), 256, 0, stream>>>(H1, W2, Xt2);
  // 8) Xe2 = segsum_h(Xt2[n])
  scatter3_k<<<cdiv(N_INC_, 256), 256, 0, stream>>>(ei0, ei1, Xt2, nullptr, Xe2);
  // 9) H2 = segsum_n(Xe2[h] * Binv[h]); then * 1/D + b2
  scatter3_k<<<cdiv(N_INC_, 256), 256, 0, stream>>>(ei1, ei0, Xe2, Bdeg, H2);
  finish3_k<<<cdiv(N_NODES_, 256), 256, 0, stream>>>(H2, Ddeg, b2, N_NODES_);
  // 10) XeM = segsum_h(H2[n]) (mean via 1/cnt inside hedge_final)
  scatter3_k<<<cdiv(N_INC_, 256), 256, 0, stream>>>(ei0, ei1, H2, nullptr, XeM);
  // 11) sigmoid -> rownorm -> gumbel argmax
  hedge_final_k<<<cdiv(N_HEDGES_, 256), 256, 0, stream>>>(XeM, Bdeg, gu, hard);
  // 12) sample + masked edge_index
  final_out_k<<<cdiv(N_INC_, 256), 256, 0, stream>>>(ei0, ei1, ov, hard,
                                                     out_sample, out_ei);
}
